// DenoisingAE_84731114815802
// MI455X (gfx1250) — matrix operand order repaired and checked by reading, not compiled, not run on hardware
//
#include <hip/hip_runtime.h>

// ---------------------------------------------------------------------------
// DenoisingAE seq2seq LSTM for MI455X (gfx1250).
//
// Weights converted once to f16 (~122 MB) so the full model stays resident in
// the 192 MB L2; every sequential LSTM step then streams weights from L2
// across the whole chip. One persistent kernel (32 WGs x 256 thr = 256 waves)
// with a global grid barrier between matvec/cell/logits phases. Mat-vecs use
// v_wmma_f32_16x16x32_f16 with a branch-free broadcast B operand (h-vector in
// all 16 columns; column 0 extracted). fp32 accumulate + fp32 cell math.
// ---------------------------------------------------------------------------

#define VSZ 32000
#define ESZ 512
#define HSZ 1024
#define SSZ 512
#define LSZ 256
#define NB  32      // workgroups in persistent kernel
#define NT  256     // threads per workgroup -> 8 waves -> 256 waves total

typedef __attribute__((ext_vector_type(16))) _Float16 v16h;
typedef __attribute__((ext_vector_type(8)))  _Float16 v8h;
typedef __attribute__((ext_vector_type(8)))  float    v8f;

// f16 weight arena offsets (in halves) inside d_ws at byte offset WTS_OFF
constexpr size_t O_EIH0 = 0;                                   // 4H x E
constexpr size_t O_EHH0 = O_EIH0 + (size_t)4 * HSZ * ESZ;      // 4H x H
constexpr size_t O_EIH1 = O_EHH0 + (size_t)4 * HSZ * HSZ;      // 4H x H
constexpr size_t O_EHH1 = O_EIH1 + (size_t)4 * HSZ * HSZ;      // 4H x H
constexpr size_t O_FCW  = O_EHH1 + (size_t)4 * HSZ * HSZ;      // H x H
constexpr size_t O_DHH0 = O_FCW  + (size_t)HSZ * HSZ;          // 4H x H
constexpr size_t O_DIH1 = O_DHH0 + (size_t)4 * HSZ * HSZ;      // 4H x H
constexpr size_t O_DHH1 = O_DIH1 + (size_t)4 * HSZ * HSZ;      // 4H x H
constexpr size_t O_DFC  = O_DHH1 + (size_t)4 * HSZ * HSZ;      // V x H
constexpr size_t WTS_OFF = 128 * 1024;                          // byte offset

// ---------------------------------------------------------------------------
__global__ void cvt_f32_to_f16(const float* __restrict__ src,
                               _Float16* __restrict__ dst, int n) {
  int i = blockIdx.x * blockDim.x + threadIdx.x;
  int stride = gridDim.x * blockDim.x;
  for (; i < n; i += stride) dst[i] = (_Float16)src[i];
}

// zero output row 0 and the control block (barrier counters, reduce slots)
__global__ void init_kernel(float* __restrict__ out, int* __restrict__ ctrl) {
  int i = blockIdx.x * blockDim.x + threadIdx.x;
  int stride = gridDim.x * blockDim.x;
  for (int j = i; j < VSZ; j += stride) out[j] = 0.0f;
  for (int j = i; j < 1024; j += stride) ctrl[j] = 0;   // first 4 KB of ws
}

// ---------------------------------------------------------------------------
// WMMA helpers: D(16x16) = A(16x32) * B(32x16) + C; the mat-vec result is
// read from column 0 of D.
// A layout (16-bit A 16x32, contiguous K order): lane<16 -> row M=lane,
// elems 0..15 = K{kb..kb+15}; lane>=16 -> row M=lane-16,
// elems = K{kb+16..kb+31}.
// B: the h-vector is BROADCAST into all 16 columns (branch-free): lane n<16
// holds K kb..kb+15 of column n, lane n+16 holds K kb+16..kb+31. Every lane
// therefore loads the same 32B chunk selected only by lane>=16.
// ---------------------------------------------------------------------------
__device__ __forceinline__ v16h load_a16(const _Float16* __restrict__ W,
                                         int ldw, int row0, int kb, int lane) {
  int r = row0 + (lane & 15);
  int s = (lane >> 4) << 4;                  // 0 or 16
  const _Float16* p = W + (size_t)r * ldw + kb + s;
  union { v16h v; v8h h[2]; } u;
  u.h[0] = *(const v8h*)(p);
  u.h[1] = *(const v8h*)(p + 8);
  return u.v;
}

__device__ __forceinline__ v16h load_b16(const _Float16* __restrict__ xv,
                                         int kb, int lane) {
  return *(const v16h*)(xv + kb + ((lane >> 4) << 4));   // +0 or +16
}

__device__ __forceinline__ void mv_acc(v8f& acc, const _Float16* __restrict__ W,
                                       int ldw, int row0,
                                       const _Float16* __restrict__ xv, int K,
                                       int lane) {
  v8f acc1 = {};                              // 2nd accumulator: halves the
  for (int kb = 0; kb < K; kb += 64) {        // WMMA RAW dependency chain
    v16h a0 = load_a16(W, ldw, row0, kb, lane);
    v16h b0 = load_b16(xv, kb, lane);
    acc = __builtin_amdgcn_wmma_f32_16x16x32_f16(false, a0, false, b0,
                                                 (short)0, acc, false, false);
    v16h a1 = load_a16(W, ldw, row0, kb + 32, lane);
    v16h b1 = load_b16(xv, kb + 32, lane);
    acc1 = __builtin_amdgcn_wmma_f32_16x16x32_f16(false, a1, false, b1,
                                                  (short)0, acc1, false, false);
  }
  acc = acc + acc1;
}

// Column 0 of D: lane0 holds M=row0..row0+7 in acc[0..7]; lane16 M=+8..+15.
// Two float4 stores per active lane (global_store_b128).
__device__ __forceinline__ void store_col0(float* __restrict__ dst, int row0,
                                           v8f acc, int lane) {
  if (lane == 0 || lane == 16) {
    int rb = row0 + ((lane >> 4) << 3);
    *(float4*)(dst + rb)     = make_float4(acc[0], acc[1], acc[2], acc[3]);
    *(float4*)(dst + rb + 4) = make_float4(acc[4], acc[5], acc[6], acc[7]);
  }
}

__device__ __forceinline__ float sigm(float x) {
  return 1.0f / (1.0f + __expf(-x));
}

// Global sense-reversing grid barrier (all NB workgroups co-resident).
__device__ __forceinline__ void grid_bar(int* cnt, int* gen) {
  __syncthreads();
  if (threadIdx.x == 0) {
    __threadfence();
    int g = __hip_atomic_load(gen, __ATOMIC_RELAXED, __HIP_MEMORY_SCOPE_AGENT);
    if (__hip_atomic_fetch_add(cnt, 1, __ATOMIC_ACQ_REL,
                               __HIP_MEMORY_SCOPE_AGENT) == NB - 1) {
      __hip_atomic_store(cnt, 0, __ATOMIC_RELAXED, __HIP_MEMORY_SCOPE_AGENT);
      __hip_atomic_store(gen, g + 1, __ATOMIC_RELEASE,
                         __HIP_MEMORY_SCOPE_AGENT);
    } else {
      while (__hip_atomic_load(gen, __ATOMIC_ACQUIRE,
                               __HIP_MEMORY_SCOPE_AGENT) == g) {
        __builtin_amdgcn_s_sleep(2);
      }
    }
    __threadfence();
  }
  __syncthreads();
}

// ---------------------------------------------------------------------------
// Persistent encoder + decoder kernel.
// ---------------------------------------------------------------------------
__global__ __launch_bounds__(NT, 1) void seq2seq_kernel(
    const int* __restrict__ x, const int* __restrict__ y,
    const float* __restrict__ emb,
    const float* __restrict__ e_bih0, const float* __restrict__ e_bhh0,
    const float* __restrict__ e_bih1, const float* __restrict__ e_bhh1,
    const float* __restrict__ e_fcb,
    const float* __restrict__ d_Wih0,
    const float* __restrict__ d_bih0, const float* __restrict__ d_bhh0,
    const float* __restrict__ d_bih1, const float* __restrict__ d_bhh1,
    const float* __restrict__ d_fcb,
    char* __restrict__ ws, float* __restrict__ out) {
  // control block (zeroed by init_kernel)
  int*   bar_cnt = (int*)ws;
  int*   bar_gen = (int*)ws + 1;
  float* xf      = (float*)ws + 16;
  float* red_v   = (float*)ws + 64;   // NB slots
  int*   red_i   = (int*)ws + 96;     // NB slots
  // fp32 state
  float* st = (float*)(ws + 4096);
  float* h0 = st;              float* c0 = st + HSZ;
  float* h1 = st + 2 * HSZ;    float* c1 = st + 3 * HSZ;
  float* g0 = st + 4 * HSZ;                 // 4H gate pre-activations, L0
  float* g1 = st + 4 * HSZ + 4 * HSZ;       // 4H gate pre-activations, L1
  // f16 state (operands for WMMA B)
  _Float16* hh  = (_Float16*)(ws + 4096 + 12 * HSZ * sizeof(float));
  _Float16* h0h = hh;
  _Float16* h1h = hh + HSZ;
  _Float16* eh  = hh + 2 * HSZ;             // E halves (embedded token)
  // f16 weights
  const _Float16* wts   = (const _Float16*)(ws + WTS_OFF);
  const _Float16* wEih0 = wts + O_EIH0;
  const _Float16* wEhh0 = wts + O_EHH0;
  const _Float16* wEih1 = wts + O_EIH1;
  const _Float16* wEhh1 = wts + O_EHH1;
  const _Float16* wFc   = wts + O_FCW;
  const _Float16* wDhh0 = wts + O_DHH0;
  const _Float16* wDih1 = wts + O_DIH1;
  const _Float16* wDhh1 = wts + O_DHH1;
  const _Float16* wDfc  = wts + O_DFC;

  __shared__ float sv[NT];
  __shared__ int   si[NT];

  const int lane  = threadIdx.x & 31;
  const int tid   = blockIdx.x * NT + threadIdx.x;
  const int gwave = tid >> 5;               // 0..255
  const int row0  = gwave * 16;

  // ---- init: zero states, embed token 0 ----
  if (tid < HSZ) {
    h0[tid] = 0.f; c0[tid] = 0.f; h1[tid] = 0.f; c1[tid] = 0.f;
    h0h[tid] = (_Float16)0.f; h1h[tid] = (_Float16)0.f;
  }
  if (tid < ESZ) eh[tid] = (_Float16)emb[(size_t)x[0] * ESZ + tid];
  grid_bar(bar_cnt, bar_gen);

  // ---- encoder: 512 steps, 2-layer LSTM ----
  for (int t = 0; t < SSZ; ++t) {
    { // gates0 = Wih0 @ e_t + Whh0 @ h0       (256 tiles over 256 waves)
      v8f acc = {};
      mv_acc(acc, wEih0, ESZ, row0, eh, ESZ, lane);
      mv_acc(acc, wEhh0, HSZ, row0, h0h, HSZ, lane);
      store_col0(g0, row0, acc, lane);
    }
    grid_bar(bar_cnt, bar_gen);
    if (tid < HSZ) { // cell 0
      float pi = g0[tid]           + e_bih0[tid]           + e_bhh0[tid];
      float pf = g0[HSZ + tid]     + e_bih0[HSZ + tid]     + e_bhh0[HSZ + tid];
      float pg = g0[2 * HSZ + tid] + e_bih0[2 * HSZ + tid] + e_bhh0[2 * HSZ + tid];
      float po = g0[3 * HSZ + tid] + e_bih0[3 * HSZ + tid] + e_bhh0[3 * HSZ + tid];
      float c2 = sigm(pf) * c0[tid] + sigm(pi) * tanhf(pg);
      float h2 = sigm(po) * tanhf(c2);
      c0[tid] = c2; h0[tid] = h2; h0h[tid] = (_Float16)h2;
    }
    grid_bar(bar_cnt, bar_gen);
    { // gates1 = Wih1 @ h0 + Whh1 @ h1
      v8f acc = {};
      mv_acc(acc, wEih1, HSZ, row0, h0h, HSZ, lane);
      mv_acc(acc, wEhh1, HSZ, row0, h1h, HSZ, lane);
      store_col0(g1, row0, acc, lane);
    }
    grid_bar(bar_cnt, bar_gen);
    if (tid < HSZ) { // cell 1
      float pi = g1[tid]           + e_bih1[tid]           + e_bhh1[tid];
      float pf = g1[HSZ + tid]     + e_bih1[HSZ + tid]     + e_bhh1[HSZ + tid];
      float pg = g1[2 * HSZ + tid] + e_bih1[2 * HSZ + tid] + e_bhh1[2 * HSZ + tid];
      float po = g1[3 * HSZ + tid] + e_bih1[3 * HSZ + tid] + e_bhh1[3 * HSZ + tid];
      float c2 = sigm(pf) * c1[tid] + sigm(pi) * tanhf(pg);
      float h2 = sigm(po) * tanhf(c2);
      c1[tid] = c2; h1[tid] = h2; h1h[tid] = (_Float16)h2;
    }
    if (t + 1 < SSZ && tid < ESZ)   // pre-embed next token (independent)
      eh[tid] = (_Float16)emb[(size_t)x[t + 1] * ESZ + tid];
    grid_bar(bar_cnt, bar_gen);
  }

  // ---- latent: lat = relu(fcW @ h1 + fcb) -> scratch g0 ----
  if (gwave < HSZ / 16) {
    v8f acc = {};
    mv_acc(acc, wFc, HSZ, row0, h1h, HSZ, lane);
    if (lane == 0 || lane == 16) {
      int rb = row0 + ((lane >> 4) << 3);
#pragma unroll
      for (int j = 0; j < 8; ++j) {
        float v = acc[j] + e_fcb[rb + j];
        g0[rb + j] = v > 0.f ? v : 0.f;
      }
    }
  }
  grid_bar(bar_cnt, bar_gen);
  // decoder h0 = h1 = lat (quirk: replicated; cell state re-seeded from h)
  if (tid < HSZ) {
    float v = g0[tid];
    h0[tid] = v; h1[tid] = v;
    h0h[tid] = (_Float16)v; h1h[tid] = (_Float16)v;
  }
  if (tid == 0) *xf = (float)y[0];
  grid_bar(bar_cnt, bar_gen);

  // ---- decoder: 255 free-running greedy steps ----
  for (int t = 1; t < LSZ; ++t) {
    // A: (WG0/thread0 also finalizes previous step's argmax -> xf; nobody
    //     reads xf until after this phase's barrier)
    if (t > 1 && blockIdx.x == 0 && threadIdx.x == 0) {
      float b = red_v[0]; int ix = red_i[0];
      for (int k = 1; k < NB; ++k)
        if (red_v[k] > b || (red_v[k] == b && red_i[k] < ix)) {
          b = red_v[k]; ix = red_i[k];
        }
      *xf = (float)ix;
    }
    { // g0 = dWhh0 @ h0
      v8f acc = {};
      mv_acc(acc, wDhh0, HSZ, row0, h0h, HSZ, lane);
      store_col0(g0, row0, acc, lane);
    }
    grid_bar(bar_cnt, bar_gen);
    if (tid < HSZ) { // B: cell 0, c_in = h0 (reference quirk)
      float xfv = *xf;
      float pi = g0[tid]           + d_Wih0[tid] * xfv           + d_bih0[tid]           + d_bhh0[tid];
      float pf = g0[HSZ + tid]     + d_Wih0[HSZ + tid] * xfv     + d_bih0[HSZ + tid]     + d_bhh0[HSZ + tid];
      float pg = g0[2 * HSZ + tid] + d_Wih0[2 * HSZ + tid] * xfv + d_bih0[2 * HSZ + tid] + d_bhh0[2 * HSZ + tid];
      float po = g0[3 * HSZ + tid] + d_Wih0[3 * HSZ + tid] * xfv + d_bih0[3 * HSZ + tid] + d_bhh0[3 * HSZ + tid];
      float cin = h0[tid];
      float c2 = sigm(pf) * cin + sigm(pi) * tanhf(pg);
      float h2 = sigm(po) * tanhf(c2);
      h0[tid] = h2; h0h[tid] = (_Float16)h2;
    }
    grid_bar(bar_cnt, bar_gen);
    { // C: g1 = dWih1 @ h0n + dWhh1 @ h1
      v8f acc = {};
      mv_acc(acc, wDih1, HSZ, row0, h0h, HSZ, lane);
      mv_acc(acc, wDhh1, HSZ, row0, h1h, HSZ, lane);
      store_col0(g1, row0, acc, lane);
    }
    grid_bar(bar_cnt, bar_gen);
    if (tid < HSZ) { // D: cell 1, c_in = h1
      float pi = g1[tid]           + d_bih1[tid]           + d_bhh1[tid];
      float pf = g1[HSZ + tid]     + d_bih1[HSZ + tid]     + d_bhh1[HSZ + tid];
      float pg = g1[2 * HSZ + tid] + d_bih1[2 * HSZ + tid] + d_bhh1[2 * HSZ + tid];
      float po = g1[3 * HSZ + tid] + d_bih1[3 * HSZ + tid] + d_bhh1[3 * HSZ + tid];
      float cin = h1[tid];
      float c2 = sigm(pf) * cin + sigm(pi) * tanhf(pg);
      float h2 = sigm(po) * tanhf(c2);
      h1[tid] = h2; h1h[tid] = (_Float16)h2;
    }
    grid_bar(bar_cnt, bar_gen);
    // E: logits = dfcW @ h1n + dfcb (2000 tiles over 256 waves) + argmax
    float bv = -3.4e38f;
    int   bi = 0x7fffffff;
    float* outrow = out + (size_t)t * VSZ;
    for (int tile = gwave; tile < VSZ / 16; tile += NB * NT / 32) {
      int r0 = tile * 16;
      v8f acc = {};
      mv_acc(acc, wDfc, HSZ, r0, h1h, HSZ, lane);
      if (lane == 0 || lane == 16) {
        int rb = r0 + ((lane >> 4) << 3);
        float v0 = acc[0] + d_fcb[rb + 0];
        float v1 = acc[1] + d_fcb[rb + 1];
        float v2 = acc[2] + d_fcb[rb + 2];
        float v3 = acc[3] + d_fcb[rb + 3];
        float v4 = acc[4] + d_fcb[rb + 4];
        float v5 = acc[5] + d_fcb[rb + 5];
        float v6 = acc[6] + d_fcb[rb + 6];
        float v7 = acc[7] + d_fcb[rb + 7];
        *(float4*)(outrow + rb)     = make_float4(v0, v1, v2, v3);
        *(float4*)(outrow + rb + 4) = make_float4(v4, v5, v6, v7);
        float vv[8] = {v0, v1, v2, v3, v4, v5, v6, v7};
#pragma unroll
        for (int j = 0; j < 8; ++j) {
          int r = rb + j;
          if (vv[j] > bv || (vv[j] == bv && r < bi)) { bv = vv[j]; bi = r; }
        }
      }
    }
    sv[threadIdx.x] = bv; si[threadIdx.x] = bi;
    __syncthreads();
    if (threadIdx.x == 0) { // per-WG argmax (first-max tie-break)
      float b = sv[0]; int ix = si[0];
      for (int k = 1; k < NT; ++k)
        if (sv[k] > b || (sv[k] == b && si[k] < ix)) { b = sv[k]; ix = si[k]; }
      red_v[blockIdx.x] = b; red_i[blockIdx.x] = ix;
    }
    grid_bar(bar_cnt, bar_gen);
    // grid-level argmax for this step is folded into phase A of step t+1
    // (the last step needs no feedback).
  }
}

// ---------------------------------------------------------------------------
extern "C" void kernel_launch(void* const* d_in, const int* in_sizes, int n_in,
                              void* d_out, int out_size, void* d_ws, size_t ws_size,
                              hipStream_t stream) {
  (void)in_sizes; (void)n_in; (void)out_size; (void)ws_size;
  const int*   x      = (const int*)d_in[0];
  const int*   y      = (const int*)d_in[1];
  const float* emb    = (const float*)d_in[2];
  const float* eWih0  = (const float*)d_in[3];
  const float* eWhh0  = (const float*)d_in[4];
  const float* e_bih0 = (const float*)d_in[5];
  const float* e_bhh0 = (const float*)d_in[6];
  const float* eWih1  = (const float*)d_in[7];
  const float* eWhh1  = (const float*)d_in[8];
  const float* e_bih1 = (const float*)d_in[9];
  const float* e_bhh1 = (const float*)d_in[10];
  const float* eFcW   = (const float*)d_in[11];
  const float* e_fcb  = (const float*)d_in[12];
  const float* dWih0  = (const float*)d_in[13];
  const float* dWhh0  = (const float*)d_in[14];
  const float* d_bih0 = (const float*)d_in[15];
  const float* d_bhh0 = (const float*)d_in[16];
  const float* dWih1  = (const float*)d_in[17];
  const float* dWhh1  = (const float*)d_in[18];
  const float* d_bih1 = (const float*)d_in[19];
  const float* d_bhh1 = (const float*)d_in[20];
  const float* dFcW   = (const float*)d_in[21];
  const float* d_fcb  = (const float*)d_in[22];

  char* ws = (char*)d_ws;
  _Float16* wts = (_Float16*)(ws + WTS_OFF);

  // one-pass fp32 -> f16 weight conversion into the L2-resident arena
  cvt_f32_to_f16<<<1024, 256, 0, stream>>>(eWih0, wts + O_EIH0, 4 * HSZ * ESZ);
  cvt_f32_to_f16<<<1024, 256, 0, stream>>>(eWhh0, wts + O_EHH0, 4 * HSZ * HSZ);
  cvt_f32_to_f16<<<1024, 256, 0, stream>>>(eWih1, wts + O_EIH1, 4 * HSZ * HSZ);
  cvt_f32_to_f16<<<1024, 256, 0, stream>>>(eWhh1, wts + O_EHH1, 4 * HSZ * HSZ);
  cvt_f32_to_f16<<<1024, 256, 0, stream>>>(eFcW,  wts + O_FCW,  HSZ * HSZ);
  cvt_f32_to_f16<<<1024, 256, 0, stream>>>(dWhh0, wts + O_DHH0, 4 * HSZ * HSZ);
  cvt_f32_to_f16<<<1024, 256, 0, stream>>>(dWih1, wts + O_DIH1, 4 * HSZ * HSZ);
  cvt_f32_to_f16<<<1024, 256, 0, stream>>>(dWhh1, wts + O_DHH1, 4 * HSZ * HSZ);
  cvt_f32_to_f16<<<1024, 256, 0, stream>>>(dFcW,  wts + O_DFC,  VSZ * HSZ);

  init_kernel<<<64, 256, 0, stream>>>((float*)d_out, (int*)d_ws);

  seq2seq_kernel<<<NB, NT, 0, stream>>>(
      x, y, emb,
      e_bih0, e_bhh0, e_bih1, e_bhh1, e_fcb,
      dWih0, d_bih0, d_bhh0, d_bih1, d_bhh1, d_fcb,
      ws, (float*)d_out);
}
